// OptimizedTransformerEnhancement_67164698575298
// MI455X (gfx1250) — compile-verified, matrix operand order checked by reading, not run on hardware
//
#include <hip/hip_runtime.h>
#include <hip/hip_bf16.h>
#include <math.h>

// ---------------------------------------------------------------------------
// MI455X / gfx1250 implementation.
//  - All large matmuls run on v_wmma_f32_16x16x32_bf16 (fp32 accumulate).
//  - A is (M,K) bf16 row-major, B is supplied as W (N,K) bf16 row-major so
//    WMMA fragments are two contiguous 16B loads per lane (ISA layout:
//    lanes 0-15 hold K 0..7 / 16..23, lanes 16-31 hold K 8..15 / 24..31).
//  - GEMM: the 8 waves of a block share one 64-col W panel -> W K-chunks are
//    cooperatively staged in LDS (double-buffered, ping-pong barrier) and the
//    A fragment is register-pipelined so global loads overlap the WMMAs.
//  - Window attention: S=q.k^T WMMA from global (shift folded into gather),
//    softmax in LDS, P and V^T staged in LDS (16B-aligned rows), P.V WMMA.
//  - Convs are a small FLOP fraction -> direct VALU kernels with BN/GELU fused.
// ---------------------------------------------------------------------------

typedef __bf16 bf16_t;
typedef __bf16 v16bf __attribute__((ext_vector_type(16)));
typedef __bf16 v8bf  __attribute__((ext_vector_type(8)));
typedef float  v8f   __attribute__((ext_vector_type(8)));

#define BN_INV 0.9999950000374997f  /* 1/sqrt(1+1e-5) */

__device__ __forceinline__ float gelu_exact(float x) {
  return 0.5f * x * (1.0f + erff(x * 0.7071067811865476f));
}

__device__ __forceinline__ v16bf load_frag(const bf16_t* row, int k0, int hi) {
  union { v16bf v; v8bf h[2]; } f;
  f.h[0] = *(const v8bf*)(row + k0 + hi * 8);
  f.h[1] = *(const v8bf*)(row + k0 + 16 + hi * 8);
  return f.v;
}

__device__ __forceinline__ v8f wmma_bf16(v16bf a, v16bf b, v8f c) {
  return __builtin_amdgcn_wmma_f32_16x16x32_bf16(false, a, false, b, (short)0, c,
                                                 false, false);
}

// ---------------------------------------------------------------------------
// Generic GEMM: D[m,n] = act((sum_k A[m,k]*W[n,k] + bias[n]) * bnS[n]*BN_INV
//                            + bnSh[n]) (+ residual[m,n])
// Block: 256 threads = 8 waves, wave w -> rows [blockIdx.y*128 + 16w, +16),
// cols [blockIdx.x*64, +64) as 4 WMMA tiles. M%128==0, N%64==0, K%32==0.
// W K-chunk (64x32) staged in LDS (double buffered); A fragment pipelined.
// ---------------------------------------------------------------------------
__global__ __launch_bounds__(256) void k_gemm_bf16(
    const bf16_t* __restrict__ A, const bf16_t* __restrict__ W,
    const float* __restrict__ bias, const float* __restrict__ bnS,
    const float* __restrict__ bnSh, const float* __restrict__ residual,
    float* __restrict__ outF, bf16_t* __restrict__ outB,
    int M, int N, int K, int applyGelu) {
  __shared__ alignas(16) bf16_t sW[2][64][40];  // 32 K + pad to 80B rows

  const int lane = threadIdx.x & 31;
  const int wave = threadIdx.x >> 5;
  const int r = lane & 15, hi = lane >> 4;
  const int mBase = blockIdx.y * 128 + wave * 16;
  const int nBase = blockIdx.x * 64;

  const bf16_t* Arow = A + (size_t)(mBase + r) * K;

  // Cooperative W staging: thread -> (row 0..63, 8-elem column group).
  const int sRow = threadIdx.x >> 2;       // 0..63
  const int sCol = (threadIdx.x & 3) * 8;  // 0,8,16,24
  const bf16_t* Wsrc = W + (size_t)(nBase + sRow) * K + sCol;

  // Prologue: stage K-chunk 0, load A fragment 0.
  *(v8bf*)&sW[0][sRow][sCol] = *(const v8bf*)(Wsrc + 0);
  v16bf a = load_frag(Arow, 0, hi);
  v8f acc[4] = {};
  __syncthreads();

  for (int k0 = 0; k0 < K; k0 += 32) {
    const int buf = (k0 >> 5) & 1;
    v16bf a_n = a;
    if (k0 + 32 < K) {  // issue next chunk's loads before this chunk's math
      *(v8bf*)&sW[buf ^ 1][sRow][sCol] = *(const v8bf*)(Wsrc + k0 + 32);
      a_n = load_frag(Arow, k0 + 32, hi);
      if (k0 + 128 < K) __builtin_prefetch(Arow + k0 + 128, 0, 1);
    }
#pragma unroll
    for (int t = 0; t < 4; ++t) {
      v16bf b = load_frag(&sW[buf][t * 16 + r][0], 0, hi);
      acc[t] = wmma_bf16(a, b, acc[t]);
    }
    a = a_n;
    __syncthreads();
  }

#pragma unroll
  for (int t = 0; t < 4; ++t) {
    const int n = nBase + t * 16 + r;
#pragma unroll
    for (int i = 0; i < 8; ++i) {
      const int m = mBase + hi * 8 + i;
      float v = acc[t][i];
      if (bias) v += bias[n];
      if (bnS) v = v * (bnS[n] * BN_INV) + (bnSh ? bnSh[n] : 0.0f);
      if (applyGelu) v = gelu_exact(v);
      const size_t idx = (size_t)m * N + n;
      if (residual) v += residual[idx];
      if (outF) outF[idx] = v;
      if (outB) outB[idx] = (bf16_t)v;
    }
  }
}

// ---------------------------------------------------------------------------
// LayerNorm over last dim (C) of (rows, C) fp32 -> bf16, one block per row.
// ---------------------------------------------------------------------------
__global__ __launch_bounds__(256) void k_layernorm_bf16(
    const float* __restrict__ x, const float* __restrict__ g,
    const float* __restrict__ b, bf16_t* __restrict__ out, int C) {
  const int row = blockIdx.x;
  const float* xr = x + (size_t)row * C;
  __shared__ float red[256];
  float s = 0.f, s2 = 0.f;
  for (int c = threadIdx.x; c < C; c += 256) {
    float v = xr[c];
    s += v;
    s2 += v * v;
  }
  red[threadIdx.x] = s;
  __syncthreads();
  for (int o = 128; o > 0; o >>= 1) {
    if (threadIdx.x < o) red[threadIdx.x] += red[threadIdx.x + o];
    __syncthreads();
  }
  const float mean = red[0] / (float)C;
  __syncthreads();
  red[threadIdx.x] = s2;
  __syncthreads();
  for (int o = 128; o > 0; o >>= 1) {
    if (threadIdx.x < o) red[threadIdx.x] += red[threadIdx.x + o];
    __syncthreads();
  }
  const float var = red[0] / (float)C - mean * mean;
  const float inv = rsqrtf(var + 1e-5f);
  bf16_t* orow = out + (size_t)row * C;
  for (int c = threadIdx.x; c < C; c += 256)
    orow[c] = (bf16_t)((xr[c] - mean) * inv * g[c] + b[c]);
}

// ---------------------------------------------------------------------------
// Window attention, one block per (window, head). H=W=64, 8x8 windows of 64
// tokens, head_dim=256. qkv is (16384, 3072) bf16 with cols [3][4][256].
// shift (0 or 4) implements jnp.roll gather; write-back to the gathered rows
// is exactly the inverse roll.
// ---------------------------------------------------------------------------
__global__ __launch_bounds__(256) void k_window_attn(
    const bf16_t* __restrict__ qkv, const float* __restrict__ rpb,
    bf16_t* __restrict__ out, int shift) {
  const int win = blockIdx.x;   // b*64 + wy*8 + wx
  const int head = blockIdx.y;  // 0..3
  const int bimg = win >> 6;
  const int wy = (win >> 3) & 7;
  const int wx = win & 7;
  const int tid = threadIdx.x;
  const int lane = tid & 31, wave = tid >> 5;
  const int r = lane & 15, hi = lane >> 4;

  __shared__ alignas(16) float  sS[64][72];   // scores (fp32)
  __shared__ alignas(16) bf16_t sP[64][72];   // softmax probs (bf16)
  __shared__ alignas(16) bf16_t sVT[256][72]; // V^T: [d][token]

  auto rowOf = [&](int tkn) -> int {
    const int gy = (wy * 8 + (tkn >> 3) + shift) & 63;
    const int gx = (wx * 8 + (tkn & 7) + shift) & 63;
    return (bimg << 12) + (gy << 6) + gx;
  };

  const size_t QOFF = (size_t)head * 256;
  const size_t KOFF = 1024 + QOFF;
  const size_t VOFF = 2048 + QOFF;

  // Stage V^T into LDS (coalesced over d = tid).
  for (int j = 0; j < 64; ++j) {
    const int m = rowOf(j);
    sVT[tid][j] = qkv[(size_t)m * 3072 + VOFF + tid];
  }

  // S = (q k^T) * hd^-0.5 + rel_pos_bias ; 16 tiles, 2 per wave.
#pragma unroll
  for (int rep = 0; rep < 2; ++rep) {
    const int tile = wave * 2 + rep;
    const int ti = tile >> 2, tj = tile & 3;
    const bf16_t* qrow = qkv + (size_t)rowOf(ti * 16 + r) * 3072 + QOFF;
    const bf16_t* krow = qkv + (size_t)rowOf(tj * 16 + r) * 3072 + KOFF;
    v8f acc = {};
    for (int k0 = 0; k0 < 256; k0 += 32) {
      v16bf a = load_frag(qrow, k0, hi);
      v16bf b = load_frag(krow, k0, hi);
      acc = wmma_bf16(a, b, acc);
    }
#pragma unroll
    for (int i = 0; i < 8; ++i) {
      const int ii = ti * 16 + hi * 8 + i;  // query token
      const int jj = tj * 16 + r;           // key token
      const int dy = (ii >> 3) - (jj >> 3) + 7;
      const int dx = (ii & 7) - (jj & 7) + 7;
      sS[ii][jj] = acc[i] * 0.0625f + rpb[(dy * 15 + dx) * 4 + head];
    }
  }
  __syncthreads();

  // Row softmax (64 rows, one thread each), write P as bf16.
  if (tid < 64) {
    float mx = -3.0e38f;
    for (int j = 0; j < 64; ++j) mx = fmaxf(mx, sS[tid][j]);
    float sum = 0.f;
    for (int j = 0; j < 64; ++j) {
      const float e = expf(sS[tid][j] - mx);
      sS[tid][j] = e;
      sum += e;
    }
    const float inv = 1.0f / sum;
    for (int j = 0; j < 64; ++j) sP[tid][j] = (bf16_t)(sS[tid][j] * inv);
  }
  __syncthreads();

  // O = P V : M=64, N=256, K=64 -> 64 tiles, 8 per wave.
#pragma unroll
  for (int rep = 0; rep < 8; ++rep) {
    const int tile = wave * 8 + rep;
    const int ti = tile >> 4, tj = tile & 15;
    const bf16_t* prow = &sP[ti * 16 + r][0];
    const bf16_t* vrow = &sVT[tj * 16 + r][0];
    v8f acc = {};
#pragma unroll
    for (int k0 = 0; k0 < 64; k0 += 32) {
      v16bf a = load_frag(prow, k0, hi);
      v16bf b = load_frag(vrow, k0, hi);
      acc = wmma_bf16(a, b, acc);
    }
#pragma unroll
    for (int i = 0; i < 8; ++i) {
      const int tok = ti * 16 + hi * 8 + i;
      const int d = tj * 16 + r;
      const int m = rowOf(tok);
      out[(size_t)m * 1024 + head * 256 + d] = (bf16_t)acc[i];
    }
  }
}

// --------------------------- elementwise / conv ----------------------------
__global__ void k_f2b(const float* __restrict__ in, bf16_t* __restrict__ out,
                      size_t n) {
  size_t i = (size_t)blockIdx.x * blockDim.x + threadIdx.x;
  if (i < n) out[i] = (bf16_t)in[i];
}

__global__ void k_nchw_to_tok_bf16(const float* __restrict__ in,
                                   bf16_t* __restrict__ out, int B, int Cc,
                                   int HW) {
  size_t idx = (size_t)blockIdx.x * blockDim.x + threadIdx.x;
  size_t total = (size_t)B * Cc * HW;
  if (idx >= total) return;
  int c = (int)(idx % Cc);
  size_t t2 = idx / Cc;
  int p = (int)(t2 % HW);
  int n = (int)(t2 / HW);
  out[idx] = (bf16_t)in[((size_t)n * Cc + c) * HW + p];
}

__global__ void k_tok_to_nchw_f32(const float* __restrict__ in,
                                  float* __restrict__ out, int B, int Cc,
                                  int HW) {
  size_t idx = (size_t)blockIdx.x * blockDim.x + threadIdx.x;
  size_t total = (size_t)B * Cc * HW;
  if (idx >= total) return;
  int p = (int)(idx % HW);
  size_t t2 = idx / HW;
  int c = (int)(t2 % Cc);
  int n = (int)(t2 / Cc);
  out[idx] = in[((size_t)n * HW + p) * Cc + c];
}

__global__ void k_conv3x3s2_bn_gelu(const float* __restrict__ in,
                                    const float* __restrict__ w,
                                    const float* __restrict__ cb,
                                    const float* __restrict__ g,
                                    const float* __restrict__ bb,
                                    float* __restrict__ out, int B, int Ci,
                                    int Hi, int Wi, int Co) {
  const int Ho = Hi / 2, Wo = Wi / 2;
  size_t total = (size_t)B * Co * Ho * Wo;
  size_t idx = (size_t)blockIdx.x * blockDim.x + threadIdx.x;
  if (idx >= total) return;
  int ox = (int)(idx % Wo);
  size_t t2 = idx / Wo;
  int oy = (int)(t2 % Ho);
  t2 /= Ho;
  int co = (int)(t2 % Co);
  int n = (int)(t2 / Co);
  float acc = cb[co];
  const int iy0 = oy * 2 - 1, ix0 = ox * 2 - 1;
  for (int ci = 0; ci < Ci; ++ci) {
    const float* ip = in + ((size_t)n * Ci + ci) * Hi * Wi;
    const float* wp = w + ((size_t)co * Ci + ci) * 9;
#pragma unroll
    for (int ky = 0; ky < 3; ++ky) {
      const int iy = iy0 + ky;
      if ((unsigned)iy >= (unsigned)Hi) continue;
#pragma unroll
      for (int kx = 0; kx < 3; ++kx) {
        const int ix = ix0 + kx;
        if ((unsigned)ix >= (unsigned)Wi) continue;
        acc += ip[(size_t)iy * Wi + ix] * wp[ky * 3 + kx];
      }
    }
  }
  out[idx] = gelu_exact(acc * (g[co] * BN_INV) + bb[co]);
}

__global__ void k_convT4x4s2_bn_gelu(const float* __restrict__ in,
                                     const float* __restrict__ w,  // (Ci,Co,4,4)
                                     const float* __restrict__ cb,
                                     const float* __restrict__ g,
                                     const float* __restrict__ bb,
                                     float* __restrict__ out, int B, int Ci,
                                     int Hi, int Wi, int Co) {
  const int Ho = Hi * 2, Wo = Wi * 2;
  size_t total = (size_t)B * Co * Ho * Wo;
  size_t idx = (size_t)blockIdx.x * blockDim.x + threadIdx.x;
  if (idx >= total) return;
  int ox = (int)(idx % Wo);
  size_t t2 = idx / Wo;
  int oy = (int)(t2 % Ho);
  t2 /= Ho;
  int co = (int)(t2 % Co);
  int n = (int)(t2 / Co);
  float acc = cb[co];
  for (int ky = (oy + 1) & 1; ky < 4; ky += 2) {
    const int iy = (oy + 1 - ky) >> 1;
    if ((unsigned)iy >= (unsigned)Hi) continue;
    for (int kx = (ox + 1) & 1; kx < 4; kx += 2) {
      const int ix = (ox + 1 - kx) >> 1;
      if ((unsigned)ix >= (unsigned)Wi) continue;
      for (int ci = 0; ci < Ci; ++ci) {
        acc += in[(((size_t)n * Ci + ci) * Hi + iy) * Wi + ix] *
               w[(((size_t)ci * Co + co) * 4 + ky) * 4 + kx];
      }
    }
  }
  out[idx] = gelu_exact(acc * (g[co] * BN_INV) + bb[co]);
}

__global__ void k_dw3x3_bn_gelu(const float* __restrict__ in,
                                const float* __restrict__ w,  // (C,1,3,3)
                                const float* __restrict__ cb,
                                const float* __restrict__ g,
                                const float* __restrict__ bb,
                                float* __restrict__ out, int B, int Cc, int H,
                                int W) {
  size_t total = (size_t)B * Cc * H * W;
  size_t idx = (size_t)blockIdx.x * blockDim.x + threadIdx.x;
  if (idx >= total) return;
  int x = (int)(idx % W);
  size_t t2 = idx / W;
  int y = (int)(t2 % H);
  t2 /= H;
  int c = (int)(t2 % Cc);
  int n = (int)(t2 / Cc);
  float acc = cb[c];
  const float* ip = in + ((size_t)n * Cc + c) * H * W;
  const float* wp = w + (size_t)c * 9;
#pragma unroll
  for (int ky = 0; ky < 3; ++ky) {
    const int iy = y + ky - 1;
    if ((unsigned)iy >= (unsigned)H) continue;
#pragma unroll
    for (int kx = 0; kx < 3; ++kx) {
      const int ix = x + kx - 1;
      if ((unsigned)ix >= (unsigned)W) continue;
      acc += ip[(size_t)iy * W + ix] * wp[ky * 3 + kx];
    }
  }
  out[idx] = gelu_exact(acc * (g[c] * BN_INV) + bb[c]);
}

__global__ void k_pw1x1_bn_res(const float* __restrict__ in,
                               const float* __restrict__ w,  // (Co,Ci)
                               const float* __restrict__ cb,
                               const float* __restrict__ g,
                               const float* __restrict__ bb,
                               const float* __restrict__ identity,
                               float* __restrict__ out, int B, int Cc, int HW) {
  size_t total = (size_t)B * Cc * HW;
  size_t idx = (size_t)blockIdx.x * blockDim.x + threadIdx.x;
  if (idx >= total) return;
  int p = (int)(idx % HW);
  size_t t2 = idx / HW;
  int co = (int)(t2 % Cc);
  int n = (int)(t2 / Cc);
  float acc = cb[co];
  const float* ip = in + (size_t)n * Cc * HW + p;
  const float* wp = w + (size_t)co * Cc;
  for (int ci = 0; ci < Cc; ++ci) acc += wp[ci] * ip[(size_t)ci * HW];
  out[idx] = identity[idx] + (acc * (g[co] * BN_INV) + bb[co]);
}

// ---------------------------------------------------------------------------
// Host orchestration
// ---------------------------------------------------------------------------
// d_in flattening (setup_inputs dict insertion order):
enum { B_LN1G = 0, B_LN1B, B_QKVW, B_QKVB, B_PROJW, B_PROJB, B_RPB,
       B_LN2G, B_LN2B, B_W1, B_B1, B_W2, B_B2 };  // per-block (13 entries)

extern "C" void kernel_launch(void* const* d_in, const int* in_sizes, int n_in,
                              void* d_out, int out_size, void* d_ws,
                              size_t ws_size, hipStream_t stream) {
  (void)in_sizes; (void)n_in; (void)out_size; (void)ws_size;

  const int IN_X = 0;
  const int IN_D0W = 1, IN_D0B = 2, IN_D0G = 3, IN_D0BB = 4;
  const int IN_D1W = 5, IN_D1B = 6, IN_D1G = 7, IN_D1BB = 8;
  const int IN_IPW = 9, IN_IPB = 10, IN_IPG = 11, IN_IPBB = 12;
  const int IN_BLK0 = 13;                    // 13 entries per block x 4
  const int IN_FNG = 65, IN_FNB = 66;
  const int IN_OPW = 67, IN_OPB = 68, IN_OPG = 69, IN_OPBB = 70;
  const int IN_U0W = 71, IN_U0B = 72, IN_U0G = 73, IN_U0BB = 74;
  const int IN_U1W = 75, IN_U1B = 76, IN_U1G = 77, IN_U1BB = 78;
  const int IN_FEDWW = 79, IN_FEDWB = 80, IN_FEG1 = 81, IN_FEB1 = 82;
  const int IN_FEPWW = 83, IN_FEPWB = 84, IN_FEG2 = 85, IN_FEB2 = 86;

  auto F = [&](int i) { return (const float*)d_in[i]; };

  const size_t NE_H0 = 8388608;     // (4,128,128,128)
  const size_t NE_H1 = 4194304;     // (4,256,64,64)
  const size_t NE_T = 16777216;     // (16384,1024)
  const size_t NE_QKV = 50331648;   // (16384,3072)
  const size_t NE_HID = 67108864;   // (16384,4096)
  const size_t NE_TOK256 = 4194304; // (16384,256)
  const size_t NE_U1 = 16777216;    // (4,64,256,256)
  const int M = 16384;

  char* cur = (char*)d_ws;
  auto alloc = [&](size_t bytes) -> char* {
    char* r = cur;
    cur += (bytes + 255) & ~(size_t)255;
    return r;
  };
  float* h0 = (float*)alloc(NE_H0 * 4);
  float* h1 = (float*)alloc(NE_H1 * 4);
  bf16_t* tok256 = (bf16_t*)alloc(NE_TOK256 * 2);
  float* t = (float*)alloc(NE_T * 4);
  bf16_t* lnb = (bf16_t*)alloc(NE_T * 2);
  bf16_t* qkvb = (bf16_t*)alloc(NE_QKV * 2);
  bf16_t* attnb = (bf16_t*)alloc(NE_T * 2);
  bf16_t* hidb = (bf16_t*)alloc(NE_HID * 2);
  float* tokout = (float*)alloc(NE_TOK256 * 4);
  bf16_t* wip = (bf16_t*)alloc((size_t)262144 * 2);
  bf16_t* wop = (bf16_t*)alloc((size_t)262144 * 2);
  bf16_t *wqkv[4], *wproj[4], *w1b[4], *w2b[4];
  for (int i = 0; i < 4; ++i) {
    wqkv[i] = (bf16_t*)alloc((size_t)3145728 * 2);
    wproj[i] = (bf16_t*)alloc((size_t)1048576 * 2);
    w1b[i] = (bf16_t*)alloc((size_t)4194304 * 2);
    w2b[i] = (bf16_t*)alloc((size_t)4194304 * 2);
  }
  // Buffer reuse after the transformer stage:
  float* g0 = h1;                // (4,256,64,64)
  float* u0out = h0;             // (4,128,128,128)
  float* u1out = (float*)qkvb;   // (4,64,256,256)
  float* fe1 = (float*)hidb;     // (4,64,256,256)

  auto f2b = [&](const float* src, bf16_t* dst, size_t n) {
    k_f2b<<<(unsigned)((n + 255) / 256), 256, 0, stream>>>(src, dst, n);
  };
  auto gemm = [&](const bf16_t* A, const bf16_t* W, const float* bias,
                  const float* bnS, const float* bnSh, const float* res,
                  float* outF, bf16_t* outB, int Mm, int Nn, int Kk, int gel) {
    dim3 grid(Nn / 64, Mm / 128);
    k_gemm_bf16<<<grid, 256, 0, stream>>>(A, W, bias, bnS, bnSh, res, outF,
                                          outB, Mm, Nn, Kk, gel);
  };

  // Weight conversions (fp32 -> bf16).
  f2b(F(IN_IPW), wip, 262144);
  f2b(F(IN_OPW), wop, 262144);
  for (int i = 0; i < 4; ++i) {
    const int bb = IN_BLK0 + 13 * i;
    f2b(F(bb + B_QKVW), wqkv[i], 3145728);
    f2b(F(bb + B_PROJW), wproj[i], 1048576);
    f2b(F(bb + B_W1), w1b[i], 4194304);
    f2b(F(bb + B_W2), w2b[i], 4194304);
  }

  // Downsample path.
  k_conv3x3s2_bn_gelu<<<(unsigned)(NE_H0 / 256), 256, 0, stream>>>(
      F(IN_X), F(IN_D0W), F(IN_D0B), F(IN_D0G), F(IN_D0BB), h0, 4, 64, 256,
      256, 128);
  k_conv3x3s2_bn_gelu<<<(unsigned)(NE_H1 / 256), 256, 0, stream>>>(
      h0, F(IN_D1W), F(IN_D1B), F(IN_D1G), F(IN_D1BB), h1, 4, 128, 128, 128,
      256);
  k_nchw_to_tok_bf16<<<(unsigned)(NE_TOK256 / 256), 256, 0, stream>>>(
      h1, tok256, 4, 256, 4096);
  // ip: 1x1 conv as GEMM + BN + GELU -> t (fp32 tokens)
  gemm(tok256, wip, F(IN_IPB), F(IN_IPG), F(IN_IPBB), nullptr, t, nullptr, M,
       1024, 256, 1);

  // Transformer blocks.
  for (int i = 0; i < 4; ++i) {
    const int bb = IN_BLK0 + 13 * i;
    const int shift = (i & 1) ? 4 : 0;
    k_layernorm_bf16<<<M, 256, 0, stream>>>(t, F(bb + B_LN1G), F(bb + B_LN1B),
                                            lnb, 1024);
    gemm(lnb, wqkv[i], F(bb + B_QKVB), nullptr, nullptr, nullptr, nullptr,
         qkvb, M, 3072, 1024, 0);
    k_window_attn<<<dim3(256, 4), 256, 0, stream>>>(qkvb, F(bb + B_RPB), attnb,
                                                    shift);
    gemm(attnb, wproj[i], F(bb + B_PROJB), nullptr, nullptr, t, t, nullptr, M,
         1024, 1024, 0);
    k_layernorm_bf16<<<M, 256, 0, stream>>>(t, F(bb + B_LN2G), F(bb + B_LN2B),
                                            lnb, 1024);
    gemm(lnb, w1b[i], F(bb + B_B1), nullptr, nullptr, nullptr, nullptr, hidb,
         M, 4096, 1024, 1);
    gemm(hidb, w2b[i], F(bb + B_B2), nullptr, nullptr, t, t, nullptr, M, 1024,
         4096, 0);
  }

  // Final norm + op 1x1 conv (GEMM + BN + GELU), back to NCHW.
  k_layernorm_bf16<<<M, 256, 0, stream>>>(t, F(IN_FNG), F(IN_FNB), lnb, 1024);
  gemm(lnb, wop, F(IN_OPB), F(IN_OPG), F(IN_OPBB), nullptr, tokout, nullptr, M,
       256, 1024, 1);
  k_tok_to_nchw_f32<<<(unsigned)(NE_TOK256 / 256), 256, 0, stream>>>(
      tokout, g0, 4, 256, 4096);

  // Upsample path.
  k_convT4x4s2_bn_gelu<<<(unsigned)(NE_H0 / 256), 256, 0, stream>>>(
      g0, F(IN_U0W), F(IN_U0B), F(IN_U0G), F(IN_U0BB), u0out, 4, 256, 64, 64,
      128);
  k_convT4x4s2_bn_gelu<<<(unsigned)(NE_U1 / 256), 256, 0, stream>>>(
      u0out, F(IN_U1W), F(IN_U1B), F(IN_U1G), F(IN_U1BB), u1out, 4, 128, 128,
      128, 64);

  // Feature-enhance: depthwise + pointwise + residual.
  k_dw3x3_bn_gelu<<<(unsigned)(NE_U1 / 256), 256, 0, stream>>>(
      u1out, F(IN_FEDWW), F(IN_FEDWB), F(IN_FEG1), F(IN_FEB1), fe1, 4, 64, 256,
      256);
  k_pw1x1_bn_res<<<(unsigned)(NE_U1 / 256), 256, 0, stream>>>(
      fe1, F(IN_FEPWW), F(IN_FEPWB), F(IN_FEG2), F(IN_FEB2), F(IN_X),
      (float*)d_out, 4, 64, 65536);
}
